// PredictionHead_43087111913923
// MI455X (gfx1250) — matrix operand order repaired and checked by reading, not compiled, not run on hardware
//
#include <hip/hip_runtime.h>

#define QN   8192
#define QDIM 256
#define HID  512
#define HH   256
#define WWI  256
#define FF   256

typedef __attribute__((ext_vector_type(16))) __bf16       v16bf;
typedef __attribute__((ext_vector_type(8)))  float        v8f;
typedef __attribute__((ext_vector_type(8)))  unsigned int v8u;

__device__ __forceinline__ unsigned int bf16rte(float f) {
  unsigned int u = __float_as_uint(f);
  return (u + 0x7FFFu + ((u >> 16) & 1u)) >> 16;
}
__device__ __forceinline__ unsigned int pkbf(float lo, float hi) {
  return bf16rte(lo) | (bf16rte(hi) << 16);
}

// ---- stage 0: repack w1 (f32 [F,HID]) into bf16 WMMA B-fragment layout ----
// fragment(n,k): 32 lanes x 8 dwords; lane<16: N=lane, K=k*32+{0..15};
// lane>=16: N=lane-16, K=k*32+16+{0..15}; dword j packs K=2j (lo), 2j+1 (hi).
__global__ __launch_bounds__(32) void prep_w1_frag(const float* __restrict__ w1,
                                                   unsigned int* __restrict__ frag) {
  const int nk   = blockIdx.x;                 // n*8 + k
  const int lane = threadIdx.x;
  const int N    = (nk >> 3) * 16 + (lane & 15);
  const int kb   = (nk & 7) * 32 + ((lane & 16) ? 16 : 0);
  unsigned int d[8];
#pragma unroll
  for (int j = 0; j < 8; ++j)
    d[j] = pkbf(w1[(kb + 2 * j) * HID + N], w1[(kb + 2 * j + 1) * HID + N]);
  uint4* p = reinterpret_cast<uint4*>(frag + (size_t)(nk * 32 + lane) * 8);
  p[0] = make_uint4(d[0], d[1], d[2], d[3]);
  p[1] = make_uint4(d[4], d[5], d[6], d[7]);
}

// ---- stage 1: v_q[hid] = sum_d w2[hid][d] * q[d]  (fp32, register-blocked) ----
__global__ __launch_bounds__(256) void vq_kernel(const float* __restrict__ queries,
                                                 const float* __restrict__ w2,
                                                 float* __restrict__ vq) {
  const int tid = threadIdx.x;
  const int qi  = blockIdx.x * 4 + (tid >> 6);
  const int hb  = (tid & 63) * 8;
  const float* qp = queries + (size_t)qi * QDIM;
  float acc[8] = {0.f, 0.f, 0.f, 0.f, 0.f, 0.f, 0.f, 0.f};
  for (int d = 0; d < QDIM; d += 4) {
    const float4 qv = *reinterpret_cast<const float4*>(qp + d);
#pragma unroll
    for (int j = 0; j < 8; ++j) {
      const float4 wv = *reinterpret_cast<const float4*>(w2 + (size_t)(hb + j) * QDIM + d);
      acc[j] += qv.x * wv.x + qv.y * wv.y + qv.z * wv.z + qv.w * wv.w;
    }
  }
#pragma unroll
  for (int j = 0; j < 8; ++j) vq[(size_t)qi * HID + hb + j] = acc[j];
}

// ---- stage 1b: c_q = q . b2 ----
__global__ __launch_bounds__(256) void cq_kernel(const float* __restrict__ queries,
                                                 const float* __restrict__ b2,
                                                 float* __restrict__ cq) {
  const int qi = blockIdx.x * 256 + threadIdx.x;
  const float* qp = queries + (size_t)qi * QDIM;
  float s = 0.f;
  for (int d = 0; d < QDIM; d += 4) {
    const float4 a = *reinterpret_cast<const float4*>(qp + d);
    const float4 b = *reinterpret_cast<const float4*>(b2 + d);
    s += a.x * b.x + a.y * b.y + a.z * b.z + a.w * b.w;
  }
  cq[qi] = s;
}

// ---- stage 2: per-query WMMA GEMM + fused relu/contract ----
__global__ __launch_bounds__(256) void head_kernel(
    const int* __restrict__ indices,
    const float* __restrict__ feats,
    const unsigned int* __restrict__ w1frag,
    const float* __restrict__ bias1,
    const float* __restrict__ vq,
    const float* __restrict__ cq,
    float* __restrict__ out) {
  const int lane = threadIdx.x & 31;
  const int qi   = blockIdx.x * 8 + (threadIdx.x >> 5);

  const int bi = indices[qi * 3 + 0];
  const int ri = indices[qi * 3 + 1];
  const int ci = indices[qi * 3 + 2];

  // window rows this lane owns (A rows 0..24 valid; 25..31 zero)
  const float* rowp[2];
  bool rowv[2];
#pragma unroll
  for (int m = 0; m < 2; ++m) {
    const int p  = m * 16 + (lane & 15);
    const int dr = p / 5 - 2, dc = p % 5 - 2;
    const int r = ri + dr, c = ci + dc;
    const bool pad = (p >= 25) || (r < 0) || (r >= HH) || (c < 0) || (c >= WWI);
    const int rc = min(max(r, 0), HH - 1);
    const int cc = min(max(c, 0), WWI - 1);
    rowp[m] = feats + (((size_t)bi * HH + rc) * WWI + cc) * FF;
    rowv[m] = !pad;
  }
  // A-fragment K layout: lanes 0-15 -> K base+0..15; lanes 16-31 -> K base+16..31
  const int kboff = (lane & 16) ? 16 : 0;

  v16bf A[8][2];
#pragma unroll
  for (int kk = 0; kk < 8; ++kk) {
#pragma unroll
    for (int m = 0; m < 2; ++m) {
      v8u d = {0u, 0u, 0u, 0u, 0u, 0u, 0u, 0u};
      if (rowv[m]) {
        const float* p = rowp[m] + kk * 32 + kboff;
        const float4 f0 = *reinterpret_cast<const float4*>(p);
        const float4 f1 = *reinterpret_cast<const float4*>(p + 4);
        const float4 f2 = *reinterpret_cast<const float4*>(p + 8);
        const float4 f3 = *reinterpret_cast<const float4*>(p + 12);
        d[0] = pkbf(f0.x, f0.y); d[1] = pkbf(f0.z, f0.w);
        d[2] = pkbf(f1.x, f1.y); d[3] = pkbf(f1.z, f1.w);
        d[4] = pkbf(f2.x, f2.y); d[5] = pkbf(f2.z, f2.w);
        d[6] = pkbf(f3.x, f3.y); d[7] = pkbf(f3.z, f3.w);
      }
      A[kk][m] = __builtin_bit_cast(v16bf, d);
    }
  }

  float acc[2][8];
#pragma unroll
  for (int m = 0; m < 2; ++m)
#pragma unroll
    for (int r = 0; r < 8; ++r) acc[m][r] = 0.0f;

  const float* vqp = vq + (size_t)qi * HID;
  const int nl = lane & 15;
  const uint4* fb = reinterpret_cast<const uint4*>(w1frag);

  for (int n = 0; n < 32; ++n) {
    v8f C0 = {0.f, 0.f, 0.f, 0.f, 0.f, 0.f, 0.f, 0.f};
    v8f C1 = {0.f, 0.f, 0.f, 0.f, 0.f, 0.f, 0.f, 0.f};
#pragma unroll
    for (int kk = 0; kk < 8; ++kk) {
      const size_t idx = ((size_t)(n * 8 + kk) * 32 + lane) * 2;
      const uint4 w0 = fb[idx];
      const uint4 w1v = fb[idx + 1];
      v8u bu = {w0.x, w0.y, w0.z, w0.w, w1v.x, w1v.y, w1v.z, w1v.w};
      const v16bf Bf = __builtin_bit_cast(v16bf, bu);
      C0 = __builtin_amdgcn_wmma_f32_16x16x32_bf16(false, A[kk][0], false, Bf,
                                                   (short)0, C0, false, false);
      C1 = __builtin_amdgcn_wmma_f32_16x16x32_bf16(false, A[kk][1], false, Bf,
                                                   (short)0, C1, false, false);
    }
    const int Ng = n * 16 + nl;
    const float bv = bias1[Ng];
    const float vv = vqp[Ng];
#pragma unroll
    for (int r = 0; r < 8; ++r) {
      float h0 = C0[r] + bv; h0 = h0 > 0.f ? h0 : 0.f;
      float h1 = C1[r] + bv; h1 = h1 > 0.f ? h1 : 0.f;
      acc[0][r] += h0 * vv;
      acc[1][r] += h1 * vv;
    }
  }

  // sum the 16 lanes that share each output row (halves stay disjoint, masks<16)
#pragma unroll
  for (int m = 0; m < 2; ++m)
#pragma unroll
    for (int r = 0; r < 8; ++r) {
      float v = acc[m][r];
      v += __shfl_xor(v, 1, 32);
      v += __shfl_xor(v, 2, 32);
      v += __shfl_xor(v, 4, 32);
      v += __shfl_xor(v, 8, 32);
      acc[m][r] = v;
    }

  const float cqv = cq[qi];
  if ((lane & 15) == 0) {           // lane 0: rows 16m+r, lane 16: rows 16m+8+r
    const int half = (lane >> 4) * 8;
#pragma unroll
    for (int m = 0; m < 2; ++m)
#pragma unroll
      for (int r = 0; r < 8; ++r) {
        const int p = m * 16 + half + r;
        if (p < 25) {
          const int dr = p / 5 - 2, dc = p % 5 - 2;
          const int rr = ri + dr, cc = ci + dc;
          const bool pad = (rr < 0) || (rr >= HH) || (cc < 0) || (cc >= WWI);
          out[(size_t)qi * 25 + p] = pad ? 0.0f : (acc[m][r] + cqv);
        }
      }
  }
}

extern "C" void kernel_launch(void* const* d_in, const int* in_sizes, int n_in,
                              void* d_out, int out_size, void* d_ws, size_t ws_size,
                              hipStream_t stream) {
  (void)in_sizes; (void)n_in; (void)out_size; (void)ws_size;
  const float* queries = (const float*)d_in[0];
  const int*   indices = (const int*)d_in[1];
  const float* feats   = (const float*)d_in[2];
  const float* w1      = (const float*)d_in[3];
  const float* b1      = (const float*)d_in[4];
  const float* w2      = (const float*)d_in[5];
  const float* b2      = (const float*)d_in[6];
  float* out = (float*)d_out;

  char* ws = (char*)d_ws;
  float* vq = (float*)ws;                                   // QN*HID f32 (16 MiB)
  float* cq = (float*)(ws + (size_t)QN * HID * 4);          // QN f32
  unsigned int* w1frag =
      (unsigned int*)(ws + (size_t)QN * HID * 4 + (size_t)QN * 4);  // 256 KiB

  prep_w1_frag<<<256, 32, 0, stream>>>(w1, w1frag);
  vq_kernel<<<QN / 4, 256, 0, stream>>>(queries, w2, vq);
  cq_kernel<<<QN / 256, 256, 0, stream>>>(queries, b2, cq);
  head_kernel<<<QN / 8, 256, 0, stream>>>(indices, feats, w1frag, b1, vq, cq, out);
}
